// GNN_predict_24051816858302
// MI455X (gfx1250) — matrix operand order repaired and checked by reading, not compiled, not run on hardware
//
#include <hip/hip_runtime.h>
#include <hip/hip_bf16.h>

typedef __attribute__((ext_vector_type(16))) _Float16 v16h;
typedef __attribute__((ext_vector_type(8)))  _Float16 v8h;
typedef __attribute__((ext_vector_type(8)))  float    v8f;

#define WPB 8     // waves per block
#define TPB 256   // 8 waves * 32 lanes (wave32)

// ==================== packing helpers ====================

__device__ __forceinline__ float fetch_seg(int r, int c,
    const float* __restrict__ s0, int w0,
    const float* __restrict__ s1, int w1,
    const float* __restrict__ s2, int w2) {
  if (c < w0) return s0[r * w0 + c];
  c -= w0;
  if (c < w1) return s1[r * w1 + c];
  c -= w1;
  if (c < w2) return s2[r * w2 + c];
  return 0.0f;
}

// X[r][c] (f16, zero-padded to Kpad) = concat(s0,s1,s2)[r][c]
__global__ void pack_kernel(const float* __restrict__ s0, int w0,
                            const float* __restrict__ s1, int w1,
                            const float* __restrict__ s2, int w2,
                            _Float16* __restrict__ X, int Kpad, int K, int rows) {
  int i = blockIdx.x * blockDim.x + threadIdx.x;
  if (i >= rows * Kpad) return;
  int r = i / Kpad;
  int c = i - r * Kpad;
  float v = (c < K) ? fetch_seg(r, c, s0, w0, s1, w1, s2, w2) : 0.0f;
  X[i] = (_Float16)v;
}

// Wh[n][k] (f16, [32][Kpad], zero-padded) = W[n][k] (f32, [OUT][K])
__global__ void wcvt_kernel(const float* __restrict__ W, int OUT, int K,
                            _Float16* __restrict__ Wh, int Kpad) {
  int i = blockIdx.x * blockDim.x + threadIdx.x;
  if (i >= 32 * Kpad) return;
  int n = i / Kpad;
  int k = i - n * Kpad;
  Wh[i] = (n < OUT && k < K) ? (_Float16)W[n * K + k] : (_Float16)0.0f;
}

// H[r][c] (f16, [N][32]) = prelu(tr[r][c], a) for c<30, else 0
__global__ void prelu_pack_kernel(const float* __restrict__ tr,
                                  const float* __restrict__ aPtr,
                                  _Float16* __restrict__ H, int N) {
  int i = blockIdx.x * blockDim.x + threadIdx.x;
  if (i >= N * 32) return;
  int r = i >> 5;
  int c = i & 31;
  float v = 0.0f;
  if (c < 30) {
    v = tr[r * 30 + c];
    float a = *aPtr;
    if (v < 0.0f) v *= a;
  }
  H[i] = (_Float16)v;
}

// ==================== WMMA GEMM on packed f16 inputs ====================
// out[r, ocol0+n] = act( sum_k X[r,k] * Wh[n,k] + bias[n] )
template <bool TWO>
__global__ void gemm_wmma_kernel(
    const _Float16* __restrict__ X, int Kpad,
    const _Float16* __restrict__ Wh,       // [32][Kpad], zero-padded
    const float* __restrict__ bias,
    const float* __restrict__ alphaPtr, int applyPrelu,
    float* __restrict__ outF, _Float16* __restrict__ outH,
    int ldo, int ocol0, int OUT, int nrows)
{
  const int lane  = threadIdx.x & 31;
  const int wave  = threadIdx.x >> 5;
  const int tile  = blockIdx.x * WPB + wave;
  if (tile * 16 >= nrows) return;              // wave-uniform
  const int khalf = lane >> 4;
  const int lrow  = lane & 15;
  const int row   = tile * 16 + lrow;

  v8f c0 = {}; v8f c1 = {};
  const int ksteps = Kpad >> 5;
  const _Float16* xrow = X + (size_t)row * Kpad + 16 * khalf;
  const _Float16* wr0  = Wh + (size_t)lrow * Kpad + 16 * khalf;
  const _Float16* wr1  = Wh + (size_t)(16 + lrow) * Kpad + 16 * khalf;

  for (int ks = 0; ks < ksteps; ++ks) {
    // Issue all loads for this k-step together (one wait, two back-to-back WMMAs)
    v8h alo = *(const v8h*)(xrow + ks * 32);
    v8h ahi = *(const v8h*)(xrow + ks * 32 + 8);
    v8h b0l = *(const v8h*)(wr0 + ks * 32);
    v8h b0h = *(const v8h*)(wr0 + ks * 32 + 8);
    v8h b1l, b1h;
    if (TWO) {
      b1l = *(const v8h*)(wr1 + ks * 32);
      b1h = *(const v8h*)(wr1 + ks * 32 + 8);
    }
    v16h a, b0, b1;
#pragma unroll
    for (int j = 0; j < 8; ++j) { a[j] = alo[j]; a[8 + j] = ahi[j]; }
#pragma unroll
    for (int j = 0; j < 8; ++j) { b0[j] = b0l[j]; b0[8 + j] = b0h[j]; }
    c0 = __builtin_amdgcn_wmma_f32_16x16x32_f16(false, a, false, b0, (short)0, c0, false, false);
    if (TWO) {
#pragma unroll
      for (int j = 0; j < 8; ++j) { b1[j] = b1l[j]; b1[8 + j] = b1h[j]; }
      c1 = __builtin_amdgcn_wmma_f32_16x16x32_f16(false, a, false, b1, (short)0, c1, false, false);
    }
  }

  const float alpha = alphaPtr ? *alphaPtr : 0.0f;
  const float bias0 = (lrow < OUT) ? bias[lrow] : 0.0f;
  const float bias1 = (TWO && 16 + lrow < OUT) ? bias[16 + lrow] : 0.0f;
#pragma unroll
  for (int v = 0; v < 8; ++v) {
    int m = v + 8 * khalf;
    int r = tile * 16 + m;
    if (r >= nrows) continue;
    {
      int n = lrow;
      if (n < OUT) {
        float val = c0[v] + bias0;
        if (applyPrelu && val < 0.0f) val *= alpha;
        if (outH) outH[(size_t)r * ldo + ocol0 + n] = (_Float16)val;
        else      outF[(size_t)r * ldo + ocol0 + n] = val;
      }
    }
    if (TWO) {
      int n = 16 + lrow;
      if (n < OUT) {
        float val = c1[v] + bias1;
        if (applyPrelu && val < 0.0f) val *= alpha;
        if (outH) outH[(size_t)r * ldo + ocol0 + n] = (_Float16)val;
        else      outF[(size_t)r * ldo + ocol0 + n] = val;
      }
    }
  }
}

// ==================== Edge propagation: gather -> WMMA -> atomic scatter =======
// msg = prelu(Wm[30,34] @ concat(H[e0], pos[map[e0]]-pos[map[e1]]) + bm, am)
// accum[e1] += msg
__global__ void edge_prop_wmma_kernel(
    const _Float16* __restrict__ H,               // [N][32] packed, prelu'd
    const int* __restrict__ e0, const int* __restrict__ e1, int E,
    const float* __restrict__ pos, const int* __restrict__ map,
    const _Float16* __restrict__ WhM,             // [32][64] f16 padded merge wts
    const float* __restrict__ bm, const float* __restrict__ amPtr,
    float* __restrict__ accum)                    // [N,30], pre-zeroed
{
  const int lane  = threadIdx.x & 31;
  const int wave  = threadIdx.x >> 5;
  const int tile  = blockIdx.x * WPB + wave;
  if (tile * 16 >= E) return;
  const int khalf = lane >> 4;
  const int lrow  = lane & 15;
  int e = tile * 16 + lrow; if (e >= E) e = E - 1;
  const int s  = e0[e];
  const int d1 = e1[e];
  float rel0, rel1, rel2, rel3;
  {
    int ms = map[s], md = map[d1];
    rel0 = pos[ms * 4 + 0] - pos[md * 4 + 0];
    rel1 = pos[ms * 4 + 1] - pos[md * 4 + 1];
    rel2 = pos[ms * 4 + 2] - pos[md * 4 + 2];
    rel3 = pos[ms * 4 + 3] - pos[md * 4 + 3];
  }

  v8f c0 = {}; v8f c1 = {};
  const _Float16* wr0 = WhM + (size_t)lrow * 64 + 16 * khalf;
  const _Float16* wr1 = WhM + (size_t)(16 + lrow) * 64 + 16 * khalf;

  // ---- k-step 0: K = 0..31 (cols 0..29 node features, 30..31 = rel0,rel1) ----
  {
    const _Float16* hr = H + (size_t)s * 32 + 16 * khalf;
    v8h alo = *(const v8h*)hr;
    v8h ahi = *(const v8h*)(hr + 8);
    v8h b0l = *(const v8h*)(wr0);
    v8h b0h = *(const v8h*)(wr0 + 8);
    v8h b1l = *(const v8h*)(wr1);
    v8h b1h = *(const v8h*)(wr1 + 8);
    v16h a, b0, b1;
#pragma unroll
    for (int j = 0; j < 8; ++j) { a[j] = alo[j]; a[8 + j] = ahi[j]; }
    // khalf==1 lanes own K=30,31 at halves 14,15 (H has zeros there)
    a[14] = khalf ? (_Float16)rel0 : a[14];
    a[15] = khalf ? (_Float16)rel1 : a[15];
#pragma unroll
    for (int j = 0; j < 8; ++j) { b0[j] = b0l[j]; b0[8 + j] = b0h[j]; }
    c0 = __builtin_amdgcn_wmma_f32_16x16x32_f16(false, a, false, b0, (short)0, c0, false, false);
#pragma unroll
    for (int j = 0; j < 8; ++j) { b1[j] = b1l[j]; b1[8 + j] = b1h[j]; }
    c1 = __builtin_amdgcn_wmma_f32_16x16x32_f16(false, a, false, b1, (short)0, c1, false, false);
  }
  // ---- k-step 1: K = 32..63 (only K=32,33 = rel2,rel3; khalf==0 halves 0,1) ----
  {
    v8h b0l = *(const v8h*)(wr0 + 32);
    v8h b0h = *(const v8h*)(wr0 + 32 + 8);
    v8h b1l = *(const v8h*)(wr1 + 32);
    v8h b1h = *(const v8h*)(wr1 + 32 + 8);
    v16h a = {}, b0, b1;
    a[0] = khalf ? (_Float16)0.0f : (_Float16)rel2;
    a[1] = khalf ? (_Float16)0.0f : (_Float16)rel3;
#pragma unroll
    for (int j = 0; j < 8; ++j) { b0[j] = b0l[j]; b0[8 + j] = b0h[j]; }
    c0 = __builtin_amdgcn_wmma_f32_16x16x32_f16(false, a, false, b0, (short)0, c0, false, false);
#pragma unroll
    for (int j = 0; j < 8; ++j) { b1[j] = b1l[j]; b1[8 + j] = b1h[j]; }
    c1 = __builtin_amdgcn_wmma_f32_16x16x32_f16(false, a, false, b1, (short)0, c1, false, false);
  }

  const float am = *amPtr;
  const float bm0 = bm[lrow];
  const float bm1 = (16 + lrow < 30) ? bm[16 + lrow] : 0.0f;
#pragma unroll
  for (int v = 0; v < 8; ++v) {
    int m  = v + 8 * khalf;
    int em = tile * 16 + m;
    if (em >= E) continue;
    int dm = __shfl(d1, m, 32);   // lane m holds row m's destination index
    {
      int n = lrow;
      float val = c0[v] + bm0;
      if (val < 0.0f) val *= am;
      atomicAdd(&accum[(size_t)dm * 30 + n], val);
    }
    {
      int n = 16 + lrow;
      if (n < 30) {
        float val = c1[v] + bm1;
        if (val < 0.0f) val *= am;
        atomicAdd(&accum[(size_t)dm * 30 + n], val);
      }
    }
  }
}

// ==================== small elementwise kernels ====================

__global__ void rel_kernel(const int* __restrict__ sta, const int* __restrict__ srcm,
                           const float* __restrict__ locs, const float* __restrict__ srcs,
                           float* __restrict__ rel, float* __restrict__ negrel, int N) {
  int i = blockIdx.x * blockDim.x + threadIdx.x;
  if (i >= N) return;
  int a = sta[i], b = srcm[i];
#pragma unroll
  for (int j = 0; j < 4; ++j) {
    float v = locs[a * 4 + j] - srcs[b * 4 + j];
    rel[i * 4 + j] = v;
    negrel[i * 4 + j] = -v;
  }
}

__global__ void deg_kernel(const int* __restrict__ dst, int E, float* __restrict__ deg) {
  int i = blockIdx.x * blockDim.x + threadIdx.x;
  if (i < E) atomicAdd(&deg[dst[i]], 1.0f);
}

__global__ void inv_kernel(float* __restrict__ d, int N) {
  int i = blockIdx.x * blockDim.x + threadIdx.x;
  if (i < N) d[i] = 1.0f / fmaxf(d[i], 1.0f);
}

__global__ void scale_kernel(const float* __restrict__ acc, const float* __restrict__ inv,
                             float* __restrict__ out, int N) {
  int i = blockIdx.x * blockDim.x + threadIdx.x;
  if (i < N * 30) out[i] = acc[i] * inv[i / 30];
}

__global__ void segsum_kernel(const float* __restrict__ h, const int* __restrict__ srcIdx,
                              float* __restrict__ agg, int N) {
  int i = blockIdx.x * blockDim.x + threadIdx.x;
  if (i >= N * 30) return;
  int node = i / 30, j = i - node * 30;
  atomicAdd(&agg[(size_t)srcIdx[node] * 30 + j], h[i]);
}

// ==================== host side ====================

static inline int g1(long long n) { return (int)((n + 255) / 256); }

static void gemm(hipStream_t st,
                 const float* s0, int w0, const float* s1, int w1, const float* s2, int w2,
                 const float* W, const float* b, const float* alpha, int prelu,
                 float* outF, _Float16* outH, int ldo, int oc0, int OUT, int rows,
                 _Float16* X, _Float16* Wh) {
  int K    = w0 + w1 + w2;
  int Kpad = ((K + 31) / 32) * 32;
  wcvt_kernel<<<g1(32 * Kpad), 256, 0, st>>>(W, OUT, K, Wh, Kpad);
  pack_kernel<<<g1((long long)rows * Kpad), 256, 0, st>>>(s0, w0, s1, w1, s2, w2, X, Kpad, K, rows);
  int tiles  = (rows + 15) / 16;
  int blocks = (tiles + WPB - 1) / WPB;
  if (OUT > 16)
    gemm_wmma_kernel<true><<<blocks, TPB, 0, st>>>(X, Kpad, Wh, b, alpha, prelu,
                                                   outF, outH, ldo, oc0, OUT, rows);
  else
    gemm_wmma_kernel<false><<<blocks, TPB, 0, st>>>(X, Kpad, Wh, b, alpha, prelu,
                                                    outF, outH, ldo, oc0, OUT, rows);
}

static void edge_prop(hipStream_t st, const _Float16* H,
                      const int* e0, const int* e1, int E,
                      const float* pos, const int* map,
                      const _Float16* WhM, const float* bm, const float* am, float* accum) {
  int tiles  = (E + 15) / 16;
  int blocks = (tiles + WPB - 1) / WPB;
  edge_prop_wmma_kernel<<<blocks, TPB, 0, st>>>(H, e0, e1, E, pos, map, WhM, bm, am, accum);
}

extern "C" void kernel_launch(void* const* d_in, const int* in_sizes, int n_in,
                              void* d_out, int out_size, void* d_ws, size_t ws_size,
                              hipStream_t stream) {
  (void)n_in; (void)out_size; (void)ws_size;
  const int N    = in_sizes[0] / 10;     // 100000
  const int E    = in_sizes[2] / 2;      // 1000000
  const int NSRC = in_sizes[7] / 4;      // 1000

  const float* x    = (const float*)d_in[0];
  const float* mask = (const float*)d_in[1];
  const int* ev0 = (const int*)d_in[2]; const int* ev1 = ev0 + E;
  const int* cl0 = (const int*)d_in[3]; const int* cl1 = cl0 + E;
  const int* prod_src = (const int*)d_in[4] + N;   // A_src_in_product row 1
  const int* sta  = (const int*)d_in[5];           // A_src_in_sta row 0
  const int* srcm = sta + N;                       // A_src_in_sta row 1
  const float* locs = (const float*)d_in[6];
  const float* srcs = (const float*)d_in[7];
  auto PF = [&](int i) { return (const float*)d_in[i]; };

  // ---- workspace layout ----
  float* base   = (float*)d_ws;
  float* rel    = base;                      // 4N
  float* negrel = rel    + (size_t)4  * N;   // 4N
  float* mask10 = negrel + (size_t)4  * N;   // 10N
  float* tr     = mask10 + (size_t)10 * N;   // 30N
  float* tr60   = tr     + (size_t)30 * N;   // 60N (tmp20 alias pre-layers)
  float* macc   = tr60   + (size_t)60 * N;   // 30N
  float* msc    = macc   + (size_t)30 * N;   // 30N
  float* inv_ds = msc    + (size_t)30 * N;   // N
  float* inv_dc = inv_ds + (size_t)N;        // N
  float* agg    = inv_dc + (size_t)N;        // NSRC*30
  float* out15  = agg    + (size_t)NSRC*30;  // NSRC*15
  float* p30    = out15  + (size_t)NSRC*15;  // NSRC*30
  float* fend   = p30    + (size_t)NSRC*30;
  _Float16* X   = (_Float16*)fend;           // N*128 halves (GEMM input staging)
  _Float16* H16 = X   + (size_t)N * 128;     // N*32  halves (edge node features)
  _Float16* Wh  = H16 + (size_t)N * 32;      // 32*128 halves (GEMM weights)
  _Float16* WhM = Wh  + (size_t)32 * 128;    // 32*64  halves (merge weights)
  float* tmp20  = tr60;                      // alias

  // ---- relative positions, degrees ----
  rel_kernel<<<g1(N), 256, 0, stream>>>(sta, srcm, locs, srcs, rel, negrel, N);
  hipMemsetAsync(inv_ds, 0, (size_t)N * sizeof(float), stream);
  hipMemsetAsync(inv_dc, 0, (size_t)N * sizeof(float), stream);
  deg_kernel<<<g1(E), 256, 0, stream>>>(ev1, E, inv_ds);
  deg_kernel<<<g1(E), 256, 0, stream>>>(cl1, E, inv_dc);
  inv_kernel<<<g1(N), 256, 0, stream>>>(inv_ds, N);
  inv_kernel<<<g1(N), 256, 0, stream>>>(inv_dc, N);

  // ---- mask embedding: [mask|rel](14) -> 20 (prelu a) -> 10 ----
  gemm(stream, mask,10, rel,4, nullptr,0, PF(129), PF(130), PF(128), 1,
       tmp20, nullptr, 20, 0, 20, N, X, Wh);
  gemm(stream, tmp20,20, nullptr,0, nullptr,0, PF(131), PF(132), nullptr, 0,
       mask10, nullptr, 10, 0, 10, N, X, Wh);

  // ---- 5 aggregation layers ----
  for (int li = 0; li < 5; ++li) {
    int B = 8 + 24 * li;
    const float *a0 = PF(B+0), *a1 = PF(B+1), *a11 = PF(B+2), *a12 = PF(B+3),
                *a2 = PF(B+4), *a21 = PF(B+5), *a22 = PF(B+6), *am = PF(B+7);
    const float *initW = PF(B+8),  *initB = PF(B+9);
    const float *t1W   = PF(B+10), *t1B   = PF(B+11);
    const float *t2W   = PF(B+12), *t2B   = PF(B+13);
    const float *u1W   = PF(B+14), *u1B   = PF(B+15);
    const float *o1W   = PF(B+16), *o1B   = PF(B+17);
    const float *u2W   = PF(B+18), *u2B   = PF(B+19);
    const float *o2W   = PF(B+20), *o2B   = PF(B+21);
    const float *mgW   = PF(B+22), *mgB   = PF(B+23);

    // merge weights -> f16 once per layer
    wcvt_kernel<<<g1(32 * 64), 256, 0, stream>>>(mgW, 30, 34, WhM, 64);

    // init: concat(tr_in, mask10) -> 30, prelu a0
    if (li == 0)
      gemm(stream, x,10, rel,4, mask10,10, initW, initB, a0, 1,
           tr, nullptr, 30, 0, 30, N, X, Wh);
    else
      gemm(stream, tr,30, mask10,10, nullptr,0, initW, initB, a0, 1,
           tr, nullptr, 30, 0, 30, N, X, Wh);

    // block 1 / station graph
    prelu_pack_kernel<<<g1((long long)N*32), 256, 0, stream>>>(tr, a11, H16, N);
    hipMemsetAsync(macc, 0, (size_t)N * 30 * sizeof(float), stream);
    edge_prop(stream, H16, ev0, ev1, E, locs, sta, WhM, mgB, am, macc);
    scale_kernel<<<g1((long long)N*30), 256, 0, stream>>>(macc, inv_ds, msc, N);
    gemm(stream, tr,30, msc,30, mask10,10, t1W, t1B, a1, 1,
         tr60, nullptr, 60, 0, 30, N, X, Wh);

    // block 1 / source graph
    prelu_pack_kernel<<<g1((long long)N*32), 256, 0, stream>>>(tr, a12, H16, N);
    hipMemsetAsync(macc, 0, (size_t)N * 30 * sizeof(float), stream);
    edge_prop(stream, H16, cl0, cl1, E, srcs, srcm, WhM, mgB, am, macc);
    scale_kernel<<<g1((long long)N*30), 256, 0, stream>>>(macc, inv_dc, msc, N);
    gemm(stream, tr,30, msc,30, mask10,10, t2W, t2B, a1, 1,
         tr60, nullptr, 60, 30, 30, N, X, Wh);

    // block 2 / station graph (u1 written directly as packed f16 into H16)
    gemm(stream, tr60,60, nullptr,0, nullptr,0, u1W, u1B, a21, 1,
         nullptr, H16, 32, 0, 30, N, X, Wh);
    hipMemsetAsync(macc, 0, (size_t)N * 30 * sizeof(float), stream);
    edge_prop(stream, H16, ev0, ev1, E, locs, sta, WhM, mgB, am, macc);
    scale_kernel<<<g1((long long)N*30), 256, 0, stream>>>(macc, inv_ds, msc, N);
    gemm(stream, tr60,60, msc,30, mask10,10, o1W, o1B, a2, 1,
         tr, nullptr, 30, 0, 15, N, X, Wh);

    // block 2 / source graph
    gemm(stream, tr60,60, nullptr,0, nullptr,0, u2W, u2B, a22, 1,
         nullptr, H16, 32, 0, 30, N, X, Wh);
    hipMemsetAsync(macc, 0, (size_t)N * 30 * sizeof(float), stream);
    edge_prop(stream, H16, cl0, cl1, E, srcs, srcm, WhM, mgB, am, macc);
    scale_kernel<<<g1((long long)N*30), 256, 0, stream>>>(macc, inv_dc, msc, N);
    gemm(stream, tr60,60, msc,30, mask10,10, o2W, o2B, a2, 1,
         tr, nullptr, 30, 15, 15, N, X, Wh);
  }

  // ---- readout: fc1a(44->30, prelu a_in), fc1b(30->30, prelu a1), segment-add ----
  gemm(stream, tr,30, mask10,10, negrel,4, PF(141), PF(142), PF(140), 1,
       tr60, nullptr, 30, 0, 30, N, X, Wh);   // scratch: tr60 front 30N
  gemm(stream, tr60,30, nullptr,0, nullptr,0, PF(143), PF(144), PF(138), 1,
       macc, nullptr, 30, 0, 30, N, X, Wh);
  hipMemsetAsync(agg, 0, (size_t)NSRC * 30 * sizeof(float), stream);
  segsum_kernel<<<g1((long long)N*30), 256, 0, stream>>>(macc, prod_src, agg, N);

  // fc2 (30->15, prelu a2), proj l1 (15->30, prelu a), proj l2 (30->2, linear)
  gemm(stream, agg,30,   nullptr,0, nullptr,0, PF(145), PF(146), PF(139), 1,
       out15, nullptr, 15, 0, 15, NSRC, X, Wh);
  gemm(stream, out15,15, nullptr,0, nullptr,0, PF(134), PF(135), PF(133), 1,
       p30, nullptr, 30, 0, 30, NSRC, X, Wh);
  gemm(stream, p30,30,   nullptr,0, nullptr,0, PF(136), PF(137), nullptr, 0,
       (float*)d_out, nullptr, 2, 0, 2, NSRC, X, Wh);
}
